// SkipGramModel_46170898432330
// MI455X (gfx1250) — matrix operand order repaired and checked by reading, not compiled, not run on hardware
//
#include <hip/hip_runtime.h>
#include <cstdint>

// SkipGram fused SGD step for MI455X (gfx1250, wave32).
//
// Cost model: output copy of both tables (~820 MB traffic) dominates at
// ~35 us @ 23.3 TB/s HBM. Pair math is ~76 MFLOP -> noise. Strategy:
//   K1: non-temporal f32x4 streaming copy  u_weight|v_weight -> d_out
//   K2: pos pairs  (WMMA dot batch + atomic LR*grad into d_out)
//   K3: neg pairs  (same kernel, lap=0, s = -sigmoid)
// Updates are linear, so per-pair contributions go straight into d_out
// (atomic f32 add) instead of a grad buffer + apply pass.

#define EMB_SIZE 200000
#define DIM      256
#define LR       0.025f
#define LAP_NORM 0.01f

typedef __attribute__((ext_vector_type(16))) _Float16 v16h;
typedef __attribute__((ext_vector_type(8)))  float    v8f;
typedef __attribute__((ext_vector_type(4)))  float    f32x4;

// ---------------------------------------------------------------------------
// K1: stream both tables into d_out. 25.6M float4 elements total.
// Non-temporal: 410 MB working set > 192 MB L2, don't pollute it.
// ---------------------------------------------------------------------------
__global__ __launch_bounds__(256)
void copy_tables(const f32x4* __restrict__ u, const f32x4* __restrict__ v,
                 f32x4* __restrict__ out, long nPer) {
  const long stride = (long)gridDim.x * blockDim.x;
  for (long i = (long)blockIdx.x * blockDim.x + threadIdx.x; i < 2 * nPer;
       i += stride) {
    f32x4 val = (i < nPer) ? __builtin_nontemporal_load(&u[i])
                           : __builtin_nontemporal_load(&v[i - nPer]);
    __builtin_nontemporal_store(val, &out[i]);
  }
}

// ---------------------------------------------------------------------------
// K2/K3: one wave handles 16 pairs.
//   1) resolve pair -> position -> node row ids
//   2) async-copy 16 u-rows + 16 v-rows (16 x 1KB each) into LDS
//   3) 8x v_wmma_f32_16x16x32_f16 accumulate U x V^T; diagonal = dots
//   4) clip, sigmoid-LUT, then atomic LR*grad adds straight into d_out
// ---------------------------------------------------------------------------
__global__ __launch_bounds__(32)
void pair_update(const float* __restrict__ uW, const float* __restrict__ vW,
                 const int* __restrict__ nodes, const float* __restrict__ table,
                 const int* __restrict__ idxU, const int* __restrict__ idxV,
                 float* __restrict__ outU, float* __restrict__ outV,
                 float lap, int isPos, int nPairs) {
  __shared__ float uTile[16 * DIM];   // 16 KB : rows emb_u[node(idxU[r])]
  __shared__ float vTile[16 * DIM];   // 16 KB : rows emb_v[node(idxV[r])]
  __shared__ int   rowU[16];
  __shared__ int   rowV[16];
  __shared__ float sVal[16];

  const int l   = threadIdx.x;             // 0..31
  const int pb  = blockIdx.x * 16;         // first pair of this block
  const int cnt = min(16, nPairs - pb);    // pairs actually owned (exact here)

  if (l < 16) {
    int p   = min(pb + l, nPairs - 1);     // clamp keeps addresses valid
    rowU[l] = nodes[idxU[p]];
    rowV[l] = nodes[idxV[p]];
  }
  __syncthreads();

  // --- async Global->LDS staging (ASYNCcnt path, gfx1250) ---------------
  // GVS addressing: SADDR(u64 base) + VADDR(i32 byte offset); row offsets
  // max ~204.8 MB < 2^31 so i32 offsets are safe.
  const uint32_t ldsU  = (uint32_t)(uintptr_t)(void*)uTile;
  const uint32_t ldsV  = (uint32_t)(uintptr_t)(void*)vTile;
  const uint64_t baseU = (uint64_t)(uintptr_t)uW;
  const uint64_t baseV = (uint64_t)(uintptr_t)vW;
  for (int t = l; t < 16 * (DIM / 4); t += 32) {   // 1024 x 16B chunks
    const int r = t >> 6;                          // row 0..15
    const int c = t & 63;                          // 16B chunk within row
    const int offU = rowU[r] * (DIM * 4) + c * 16;
    const int offV = rowV[r] * (DIM * 4) + c * 16;
    const uint32_t du = ldsU + (uint32_t)t * 16u;
    const uint32_t dv = ldsV + (uint32_t)t * 16u;
    asm volatile("global_load_async_to_lds_b128 %0, %1, %2"
                 :: "v"(du), "v"(offU), "s"(baseU) : "memory");
    asm volatile("global_load_async_to_lds_b128 %0, %1, %2"
                 :: "v"(dv), "v"(offV), "s"(baseV) : "memory");
  }
  asm volatile("s_wait_asynccnt 0" ::: "memory");
  __syncthreads();

  // --- WMMA: acc(16x16) += A(16x32 f16) x B(32x16 f16), 8 K-chunks ------
  // A layout (ISA 16-bit A 16x32): lane<16 row=lane K {0..7,16..23};
  //                                lane>=16 row=lane-16 K {8..15,24..31}.
  // B layout: same K order as A:
  //   lanes 0-15: col=lane, K {0..7,16..23} ; lanes 16-31: col=lane-16, K {8..15,24..31}.
  const int m  = l & 15;
  const int hi = l >> 4;
  v8f acc{};
#pragma unroll
  for (int ck = 0; ck < DIM / 32; ++ck) {
    const float* ua = &uTile[m * DIM + ck * 32];
    const float* va = &vTile[m * DIM + ck * 32];
    v16h a, b;
#pragma unroll
    for (int i = 0; i < 8; ++i) {
      a[i]     = (_Float16)ua[hi * 8 + i];
      a[i + 8] = (_Float16)ua[hi * 8 + 16 + i];
    }
#pragma unroll
    for (int i = 0; i < 8; ++i) {
      b[i]     = (_Float16)va[hi * 8 + i];
      b[i + 8] = (_Float16)va[hi * 8 + 16 + i];
    }
    acc = __builtin_amdgcn_wmma_f32_16x16x32_f16(
        /*neg_a=*/false, a, /*neg_b=*/false, b,
        /*c_mod=*/(short)0, acc, /*reuse_a=*/false, /*reuse_b=*/false);
  }

  // --- diagonal extraction: (r,r) at VGPR r lane r (r<8),
  //                                  VGPR r-8 lane 24+(r-8) (r>=8) -------
  const int which = (l < 8) ? l : ((l >= 24) ? (l - 24) : -1);
  float dsc = acc[0];
#pragma unroll
  for (int i = 1; i < 8; ++i)
    if (which == i) dsc = acc[i];
  if (l < 8)        sVal[l]      = dsc;
  else if (l >= 24) sVal[l - 16] = dsc;
  __syncthreads();

  // --- clip + fast_sigmoid LUT + pos/neg coefficient --------------------
  if (l < 16) {
    float s  = fminf(fmaxf(sVal[l], -6.0f), 6.0f);
    int   ti = (int)floorf((s + 6.01f) / 0.01f);
    ti       = min(max(ti, 0), 1201);
    float sg = table[ti];
    sVal[l]  = isPos ? (1.0f - sg) : (-sg);   // NEG_WEIGHT == 1.0 folded
  }
  __syncthreads();

  // --- atomic LR*grad adds straight into the output tables --------------
  // pos: gu = s*ev + lap*(ev-eu), gv = s*eu + lap*(eu-ev);  neg: lap = 0.
#pragma unroll 1
  for (int r = 0; r < cnt; ++r) {
    const float  sr = sVal[r];
    float* du = outU + (long)rowU[r] * DIM;
    float* dv = outV + (long)rowV[r] * DIM;
    for (int d = l; d < DIM; d += 32) {
      const float eu = uTile[r * DIM + d];
      const float ev = vTile[r * DIM + d];
      const float gu = sr * ev + lap * (ev - eu);
      const float gv = sr * eu + lap * (eu - ev);
      __hip_atomic_fetch_add(&du[d], LR * gu, __ATOMIC_RELAXED,
                             __HIP_MEMORY_SCOPE_AGENT);
      __hip_atomic_fetch_add(&dv[d], LR * gv, __ATOMIC_RELAXED,
                             __HIP_MEMORY_SCOPE_AGENT);
    }
  }
}

// ---------------------------------------------------------------------------
extern "C" void kernel_launch(void* const* d_in, const int* in_sizes, int n_in,
                              void* d_out, int out_size, void* d_ws,
                              size_t ws_size, hipStream_t stream) {
  (void)n_in; (void)out_size; (void)d_ws; (void)ws_size;
  const int*   nodes = (const int*)d_in[0];
  const float* uW    = (const float*)d_in[1];
  const float* vW    = (const float*)d_in[2];
  const float* table = (const float*)d_in[3];
  const int*   posU  = (const int*)d_in[4];
  const int*   posV  = (const int*)d_in[5];
  const int*   negU  = (const int*)d_in[6];
  const int*   negV  = (const int*)d_in[7];

  float* outU = (float*)d_out;
  float* outV = outU + (long)EMB_SIZE * DIM;

  // K1: stream both tables into d_out (bandwidth floor of the whole op).
  const long nPer = (long)EMB_SIZE * DIM / 4;   // float4 count per table
  copy_tables<<<8192, 256, 0, stream>>>((const f32x4*)uW, (const f32x4*)vW,
                                        (f32x4*)d_out, nPer);

  // K2/K3: pair kernels (16 pairs per 32-thread block; counts divide by 16).
  const int nPos = in_sizes[4];   // 24640
  const int nNeg = in_sizes[6];   // 123200
  pair_update<<<(nPos + 15) / 16, 32, 0, stream>>>(
      uW, vW, nodes, table, posU, posV, outU, outV, LAP_NORM, 1, nPos);
  pair_update<<<(nNeg + 15) / 16, 32, 0, stream>>>(
      uW, vW, nodes, table, negU, negV, outU, outV, 0.0f, 0, nNeg);
}
